// PTFPiecewiseDecoder_66992899883086
// MI455X (gfx1250) — matrix operand order repaired and checked by reading, not compiled, not run on hardware
//
#include <hip/hip_runtime.h>

typedef __attribute__((ext_vector_type(16))) _Float16 v16h;
typedef __attribute__((ext_vector_type(8)))  float    v8f;

static constexpr int NJ  = 24;   // joints
static constexpr int HPG = 32;   // hidden per group
static constexpr int BNK = 4;    // bottleneck
static constexpr int CDM = 128;  // c dim
static constexpr int PD  = 3;    // point dim
static constexpr int STRIP_TILES = 16;   // 16-timestep tiles per wave

// Channel relabeling that makes the D layout of one layer feed the B operand of
// the next layer with NO cross-lane movement: swap channel ranges [8,16) <-> [16,24).
__device__ __forceinline__ int permK(int k) {
  const int g = (k >> 3) & 3;
  return (g == 1 || g == 2) ? (k ^ 24) : k;
}

// A-operand (16 x 32, f16) half index -> K:
//  lane<16 : h0..7 -> K 0..7,  h8..15 -> K 16..23
//  lane>=16: h0..7 -> K 8..15, h8..15 -> K 24..31
__device__ __forceinline__ int fragK(int h, int lane) {
  const int kbase = (lane < 16) ? 0 : 8;
  return (h < 8) ? (kbase + h) : (kbase + 8 + h);
}

// Natural-order A fragment (layer p: B operand built in natural channel order)
__device__ __forceinline__ v16h load_A_nat(const float* __restrict__ W, int Cin,
                                           int baseM, int lane) {
  const int row = baseM + (lane & 15);
  v16h a;
#pragma unroll
  for (int h = 0; h < 16; ++h) {
    const int k = fragK(h, lane);
    a[h] = (_Float16)((k < Cin) ? W[row * Cin + k] : 0.0f);
  }
  return a;
}

// Permuted A fragment (hidden layers: B operand arrives in permK channel order)
__device__ __forceinline__ v16h load_A_perm(const float* __restrict__ W,
                                            int baseM, int lane) {
  const int row = baseM + (lane & 15);
  v16h a;
#pragma unroll
  for (int h = 0; h < 16; ++h) a[h] = (_Float16)W[row * HPG + fragK(h, lane)];
  return a;
}

// Output layer as WMMA: every A row = Wout (permuted), so D row0 = the result.
__device__ __forceinline__ v16h load_A_out(const float* __restrict__ Wout_j, int lane) {
  v16h a;
#pragma unroll
  for (int h = 0; h < 16; ++h) a[h] = (_Float16)Wout_j[fragK(h, lane)];
  return a;
}

// C-operand bias fragment (C/D layout: VGPR r holds M = baseM + r + 8*(lane>=16))
__device__ __forceinline__ v8f load_biasC(const float* __restrict__ b, int baseM, int lane) {
  const int off = (lane < 16) ? 0 : 8;
  v8f cf;
#pragma unroll
  for (int r = 0; r < 8; ++r) cf[r] = b[baseM + off + r];
  return cf;
}

// ReLU(D0, D1) -> next-layer B operand in permK channel order (lane-uniform).
// Convert first (v_cvt_pk_f16_f32 x8), ReLU in f16 (v_pk_max_num_f16 x8):
// cvt(max(x,0)) == max(cvt(x), 0) since negatives convert to <= -0.
__device__ __forceinline__ v16h act_to_B(v8f d0, v8f d1) {
  v16h b;
#pragma unroll
  for (int r = 0; r < 8; ++r) {
    b[r]     = (_Float16)d0[r];
    b[8 + r] = (_Float16)d1[r];
  }
  const _Float16 z = (_Float16)0.0f;
#pragma unroll
  for (int h = 0; h < 16; ++h) b[h] = (b[h] > z) ? b[h] : z;
  return b;
}

__global__ __launch_bounds__(256) void ptf_decoder_kernel(
    const float* __restrict__ p,     const float* __restrict__ c,
    const float* __restrict__ Wproj, const float* __restrict__ bproj,
    const float* __restrict__ Wp,    const float* __restrict__ bp,
    const float* __restrict__ W0,    const float* __restrict__ b0,
    const float* __restrict__ W1,    const float* __restrict__ b1,
    const float* __restrict__ W2,    const float* __restrict__ b2,
    const float* __restrict__ Wout,  const float* __restrict__ bout,
    float* __restrict__ out,
    int B, int T, int ntiles, int nstrips, int nwaves)
{
  const int lane = threadIdx.x & 31;
  const int wave = blockIdx.x * (blockDim.x >> 5) + (threadIdx.x >> 5);
  if (wave >= nwaves) return;                 // wave-uniform guard

  const int strip = wave % nstrips;
  const int bj    = wave / nstrips;
  const int j     = bj % NJ;
  const int b     = bj / NJ;

  // ---- weights -> A fragments (VGPR-resident for the whole strip) ----
  const v16h Ap0 = load_A_nat(Wp + (size_t)j * HPG * (PD + BNK), PD + BNK, 0,  lane);
  const v16h Ap1 = load_A_nat(Wp + (size_t)j * HPG * (PD + BNK), PD + BNK, 16, lane);
  const v16h A00 = load_A_perm(W0 + (size_t)j * HPG * HPG, 0,  lane);
  const v16h A01 = load_A_perm(W0 + (size_t)j * HPG * HPG, 16, lane);
  const v16h A10 = load_A_perm(W1 + (size_t)j * HPG * HPG, 0,  lane);
  const v16h A11 = load_A_perm(W1 + (size_t)j * HPG * HPG, 16, lane);
  const v16h A20 = load_A_perm(W2 + (size_t)j * HPG * HPG, 0,  lane);
  const v16h A21 = load_A_perm(W2 + (size_t)j * HPG * HPG, 16, lane);
  const v16h Aou = load_A_out(Wout + (size_t)j * HPG, lane);

  // ---- biases -> C fragments (fused into WMMA) ----
  const v8f Cp0 = load_biasC(bp + j * HPG, 0, lane), Cp1 = load_biasC(bp + j * HPG, 16, lane);
  const v8f C00 = load_biasC(b0 + j * HPG, 0, lane), C01 = load_biasC(b0 + j * HPG, 16, lane);
  const v8f C10 = load_biasC(b1 + j * HPG, 0, lane), C11 = load_biasC(b1 + j * HPG, 16, lane);
  const v8f C20 = load_biasC(b2 + j * HPG, 0, lane), C21 = load_biasC(b2 + j * HPG, 16, lane);
  v8f Cou;
  {
    const float bo = bout[j];
#pragma unroll
    for (int r = 0; r < 8; ++r) Cou[r] = bo;
  }

  // ---- c_proj (4 scalars per (b,j)): shuffle-reduced dot products (once per wave) ----
  const int oq = lane & 3;
  const int gq = lane >> 2;                      // 8 groups of 16 channels
  const float* cb  = c + (size_t)b * CDM;
  const float* wpr = Wproj + ((size_t)j * BNK + oq) * CDM + gq * 16;
  float s = 0.0f;
#pragma unroll
  for (int k = 0; k < 16; ++k) s = fmaf(cb[gq * 16 + k], wpr[k], s);
  s += __shfl_xor(s, 4, 32);
  s += __shfl_xor(s, 8, 32);
  s += __shfl_xor(s, 16, 32);
  float cp[4];
#pragma unroll
  for (int q = 0; q < 4; ++q) cp[q] = __shfl(s, q, 32) + bproj[j * BNK + q];

  const int  t16 = lane & 15;
  const bool lo  = (lane < 16);
  const float* pj = p + ((size_t)b * (NJ * PD) + j * PD) * (size_t)T;
  float* oj = out + ((size_t)b * NJ + j) * (size_t)T;

  const int tile0 = strip * STRIP_TILES;
  const int tile1 = min(tile0 + STRIP_TILES, ntiles);

  // Build the layer-p B fragment for one tile (natural channel order):
  // K0..2 = p, K3..6 = c_proj, rest 0 (hi lanes supply K8..15,24..31 -> all zero).
  auto build_B = [&](int t) -> v16h {
    const int tc = (t < T) ? t : (T - 1);        // clamped, all lanes load (coalesced)
    const float p0 = pj[tc];
    const float p1 = pj[(size_t)T + tc];
    const float p2 = pj[2 * (size_t)T + tc];
    v16h Bx;
#pragma unroll
    for (int h = 0; h < 16; ++h) Bx[h] = (_Float16)0.0f;
    Bx[0] = (_Float16)(lo ? p0 : 0.0f);
    Bx[1] = (_Float16)(lo ? p1 : 0.0f);
    Bx[2] = (_Float16)(lo ? p2 : 0.0f);
    Bx[3] = (_Float16)(lo ? cp[0] : 0.0f);
    Bx[4] = (_Float16)(lo ? cp[1] : 0.0f);
    Bx[5] = (_Float16)(lo ? cp[2] : 0.0f);
    Bx[6] = (_Float16)(lo ? cp[3] : 0.0f);
    return Bx;
  };

  // Two tiles software-pipelined per iteration: all four WMMAs of a layer are
  // issued back-to-back, then tile A's repack runs in tile B's WMMA shadow
  // (fills the 4 co-exec hazard slots with useful work instead of v_nop).
  for (int tile = tile0; tile < tile1; tile += 2) {
    const int ta = tile * 16 + t16;
    const int tb = ta + 16;

    v16h ha = build_B(ta);
    v16h hb = build_B(tb);

    // layer p (7 -> 32)
    v8f d0a = __builtin_amdgcn_wmma_f32_16x16x32_f16(false, Ap0, false, ha, (short)0, Cp0, false, false);
    v8f d1a = __builtin_amdgcn_wmma_f32_16x16x32_f16(false, Ap1, false, ha, (short)0, Cp1, false, false);
    v8f d0b = __builtin_amdgcn_wmma_f32_16x16x32_f16(false, Ap0, false, hb, (short)0, Cp0, false, false);
    v8f d1b = __builtin_amdgcn_wmma_f32_16x16x32_f16(false, Ap1, false, hb, (short)0, Cp1, false, false);
    ha = act_to_B(d0a, d1a);
    hb = act_to_B(d0b, d1b);

    // hidden 0
    d0a = __builtin_amdgcn_wmma_f32_16x16x32_f16(false, A00, false, ha, (short)0, C00, false, false);
    d1a = __builtin_amdgcn_wmma_f32_16x16x32_f16(false, A01, false, ha, (short)0, C01, false, false);
    d0b = __builtin_amdgcn_wmma_f32_16x16x32_f16(false, A00, false, hb, (short)0, C00, false, false);
    d1b = __builtin_amdgcn_wmma_f32_16x16x32_f16(false, A01, false, hb, (short)0, C01, false, false);
    ha = act_to_B(d0a, d1a);
    hb = act_to_B(d0b, d1b);

    // hidden 1
    d0a = __builtin_amdgcn_wmma_f32_16x16x32_f16(false, A10, false, ha, (short)0, C10, false, false);
    d1a = __builtin_amdgcn_wmma_f32_16x16x32_f16(false, A11, false, ha, (short)0, C11, false, false);
    d0b = __builtin_amdgcn_wmma_f32_16x16x32_f16(false, A10, false, hb, (short)0, C10, false, false);
    d1b = __builtin_amdgcn_wmma_f32_16x16x32_f16(false, A11, false, hb, (short)0, C11, false, false);
    ha = act_to_B(d0a, d1a);
    hb = act_to_B(d0b, d1b);

    // hidden 2
    d0a = __builtin_amdgcn_wmma_f32_16x16x32_f16(false, A20, false, ha, (short)0, C20, false, false);
    d1a = __builtin_amdgcn_wmma_f32_16x16x32_f16(false, A21, false, ha, (short)0, C21, false, false);
    d0b = __builtin_amdgcn_wmma_f32_16x16x32_f16(false, A20, false, hb, (short)0, C20, false, false);
    d1b = __builtin_amdgcn_wmma_f32_16x16x32_f16(false, A21, false, hb, (short)0, C21, false, false);
    ha = act_to_B(d0a, d1a);
    hb = act_to_B(d0b, d1b);

    // output layer (32 -> 1) as a 9th WMMA: all A rows = Wout, C = b_out.
    // D row 0 (VGPR 0) holds the logit for time = lane&15 in every lane half.
    v8f dqa = __builtin_amdgcn_wmma_f32_16x16x32_f16(false, Aou, false, ha, (short)0, Cou, false, false);
    v8f dqb = __builtin_amdgcn_wmma_f32_16x16x32_f16(false, Aou, false, hb, (short)0, Cou, false, false);

    if (lo && ta < T) oj[ta] = dqa[0];
    if (lo && tb < T) oj[tb] = dqb[0];
  }
}

extern "C" void kernel_launch(void* const* d_in, const int* in_sizes, int n_in,
                              void* d_out, int out_size, void* d_ws, size_t ws_size,
                              hipStream_t stream) {
  const float* p     = (const float*)d_in[0];
  // d_in[1] = z : unused by the reference
  const float* c     = (const float*)d_in[2];
  const float* Wproj = (const float*)d_in[3];
  const float* bproj = (const float*)d_in[4];
  const float* Wp    = (const float*)d_in[5];
  const float* bp    = (const float*)d_in[6];
  const float* W0    = (const float*)d_in[7];
  const float* b0    = (const float*)d_in[8];
  const float* W1    = (const float*)d_in[9];
  const float* b1    = (const float*)d_in[10];
  const float* W2    = (const float*)d_in[11];
  const float* b2    = (const float*)d_in[12];
  const float* Wout  = (const float*)d_in[13];
  const float* bout  = (const float*)d_in[14];
  float* out = (float*)d_out;

  const int B = in_sizes[2] / CDM;                 // c is (B, 128)
  const int T = in_sizes[0] / (B * NJ * PD);       // p is (B, 72, T)
  const int ntiles  = (T + 15) / 16;
  const int nstrips = (ntiles + STRIP_TILES - 1) / STRIP_TILES;
  const int nwaves  = B * NJ * nstrips;
  const int wavesPerBlock = 8;                     // 256 threads = 8 wave32
  const int blocks = (nwaves + wavesPerBlock - 1) / wavesPerBlock;

  ptf_decoder_kernel<<<blocks, wavesPerBlock * 32, 0, stream>>>(
      p, c, Wproj, bproj, Wp, bp, W0, b0, W1, b1, W2, b2, Wout, bout,
      out, B, T, ntiles, nstrips, nwaves);
}
